// D2DUN_1846835937434
// MI455X (gfx1250) — matrix operand order repaired and checked by reading, not compiled, not run on hardware
//
#include <hip/hip_runtime.h>
#include <math.h>

// ---------------------------------------------------------------------------
// MI455X (gfx1250) implementation.
//
// Shapes: B=8, H=W=256, HW=65536.  GEMM-shaped ops (1x1 convs, dense 3x3 via
// implicit GEMM, attention @ V) run on the matrix pipe via
// v_wmma_f32_16x16x32_bf16 (bf16 A/B converted in-register, f32 accumulate).
// Roofline: ~60 GFLOP vs ~2-3 GB HBM traffic -> with WMMA the pipeline is
// HBM-bound (~100 us class at 23.3 TB/s); f32 VALU GEMMs would dominate.
//
// Operand gathers are branchless (clamped addresses + value selects) so the
// compiler emits straight-line global_load_b32 batches with one s_wait, and
// EXEC stays all-ones around every WMMA (ISA 7.12 requirement).
//
// Workspace: 11 slots x 16,777,216 floats + small attention scratch
// (~740 MB total).  Input pointer order = recursive dict order of
// setup_inputs().
// ---------------------------------------------------------------------------

#define NBATCH 8
#define IMG    256
#define HW     65536          // IMG*IMG
#define SLOT   16777216ul     // NBATCH*32*HW floats

typedef __attribute__((ext_vector_type(16))) __bf16 v16bf;
typedef __attribute__((ext_vector_type(8)))  float  v8f;

static __device__ __forceinline__ v8f wmma_bf16(v16bf a, v16bf b, v8f c) {
  return __builtin_amdgcn_wmma_f32_16x16x32_bf16(false, a, false, b, (short)0, c,
                                                 false, false);
}

static __device__ __forceinline__ float gelu_exact(float x) {
  return 0.5f * x * (1.0f + erff(x * 0.7071067811865475f));
}
static __device__ __forceinline__ float act_apply(float v, int act) {
  if (act == 1) return fmaxf(v, 0.0f);
  if (act == 2) return gelu_exact(v);
  return v;
}

// ---------------------------------------------------------------------------
// 1x1 conv as GEMM: out[b,oc,p] = act(sum_ic W[oc,ic]*in[b,ic,p] + bias) + res
// One wave -> one 16(M=oc) x 16(N=pixels) tile, K=32 steps over Cin.
// grid = (NBATCH*HW/16/8, ceil(Cout/16)), block = 256 (8 waves).
// ---------------------------------------------------------------------------
__global__ void k_conv1x1(const float* __restrict__ in, const float* __restrict__ Wt,
                          const float* __restrict__ bias, const float* __restrict__ resid,
                          float* __restrict__ out, int Cin, int Cout, int act) {
  const int lane   = threadIdx.x & 31;
  const int laneHi = lane >> 4;
  const int n      = lane & 15;
  const int wave   = blockIdx.x * (blockDim.x >> 5) + (threadIdx.x >> 5);
  const int mtile  = blockIdx.y;

  const long pix0 = (long)wave * 16;
  const int  b    = (int)(pix0 / HW);
  const int  hw   = (int)(pix0 % HW) + n;
  const int  oc_a = mtile * 16 + n;                 // A-matrix row for this lane
  const bool mFull = (mtile * 16 + 16) <= Cout;     // wave-uniform

  v8f acc = {0.f, 0.f, 0.f, 0.f, 0.f, 0.f, 0.f, 0.f};
  for (int k0 = 0; k0 < Cin; k0 += 32) {
    v16bf av, bv;
    if (mFull && (k0 + 32) <= Cin) {
      // wave-uniform fast path: no guards, constant-offset load batches
      const float* ap = Wt + (long)oc_a * Cin + k0;
      const float* bp = in + (long)(b * Cin + k0 + laneHi * 16) * HW + hw;
#pragma unroll
      for (int e = 0; e < 16; ++e) {
        av[e] = (__bf16)ap[e + laneHi * 16];
        bv[e] = (__bf16)bp[(long)e * HW];
      }
    } else {
      // tail: clamped addresses + value selects (no EXEC divergence)
      const int rowA = (oc_a < Cout) ? oc_a : (Cout - 1);
#pragma unroll
      for (int e = 0; e < 16; ++e) {
        const int kkA = k0 + e + laneHi * 16;
        const int kA  = (kkA < Cin) ? kkA : (Cin - 1);
        const float aval = Wt[(long)rowA * Cin + kA];
        av[e] = (__bf16)((kkA < Cin && oc_a < Cout) ? aval : 0.0f);
        const int kkB = k0 + e + laneHi * 16;
        const int kB  = (kkB < Cin) ? kkB : (Cin - 1);
        const float bval = in[(long)(b * Cin + kB) * HW + hw];
        bv[e] = (__bf16)((kkB < Cin) ? bval : 0.0f);
      }
    }
    acc = wmma_bf16(av, bv, acc);
  }
#pragma unroll
  for (int r = 0; r < 8; ++r) {
    const int oc = mtile * 16 + laneHi * 8 + r;
    if (oc < Cout) {
      float v = acc[r];
      if (bias) v += bias[oc];
      v = act_apply(v, act);
      const long o = (long)(b * Cout + oc) * HW + hw;
      if (resid) v += resid[o];
      out[o] = v;
    }
  }
}

// ---------------------------------------------------------------------------
// Dense 3x3 conv 32->32 (pad=1) as implicit GEMM: 9 taps x (K=32) WMMA steps.
// ---------------------------------------------------------------------------
__global__ void k_conv3x3_32(const float* __restrict__ in, const float* __restrict__ Wt,
                             const float* __restrict__ bias, float* __restrict__ out,
                             int act) {
  const int lane   = threadIdx.x & 31;
  const int laneHi = lane >> 4;
  const int n      = lane & 15;
  const int wave   = blockIdx.x * (blockDim.x >> 5) + (threadIdx.x >> 5);
  const int mtile  = blockIdx.y;   // 0..1

  const long pix0 = (long)wave * 16;
  const int  b    = (int)(pix0 / HW);
  const int  hw0  = (int)(pix0 % HW);
  const int  y    = hw0 / IMG;
  const int  x0   = hw0 % IMG;     // multiple of 16 -> tile stays in one row
  const int  oc_a = mtile * 16 + n;
  // A base for this lane: Wt[(oc_a*32 + icA)*9 + tap], icA = e+laneHi*16
  const float* apb = Wt + (long)oc_a * 288 + (long)laneHi * 16 * 9;

  v8f acc = {0.f, 0.f, 0.f, 0.f, 0.f, 0.f, 0.f, 0.f};
  for (int ky = 0; ky < 3; ++ky) {
    const int yy  = y + ky - 1;
    const int yyc = (yy < 0) ? 0 : ((yy > IMG - 1) ? IMG - 1 : yy);
    const bool okY = (yy >= 0) && (yy < IMG);
    for (int kx = 0; kx < 3; ++kx) {
      const int xx  = x0 + n + kx - 1;
      const int xxc = (xx < 0) ? 0 : ((xx > IMG - 1) ? IMG - 1 : xx);
      const bool ok = okY && (xx >= 0) && (xx < IMG);
      const int tap = ky * 3 + kx;
      const float* bp = in + (long)(b * 32 + laneHi * 16) * HW + yyc * IMG + xxc;
      v16bf av, bv;
#pragma unroll
      for (int e = 0; e < 16; ++e) {
        const int icAoff = e * 9;
        av[e] = (__bf16)apb[icAoff + tap];
        const float bval = bp[(long)e * HW];       // unconditional (clamped addr)
        bv[e] = (__bf16)(ok ? bval : 0.0f);
      }
      acc = wmma_bf16(av, bv, acc);
    }
  }
#pragma unroll
  for (int r = 0; r < 8; ++r) {
    const int oc = mtile * 16 + laneHi * 8 + r;
    float v = acc[r] + bias[oc];
    v = act_apply(v, act);
    out[(long)(b * 32 + oc) * HW + hw0 + n] = v;
  }
}

// ---------------------------------------------------------------------------
// attention @ V:  out[b,n,c,d] = sum_k att[bn,c,k] * v[b,k, n*16384+d]
// outMode 0: out[b,c, n*16384+d]        (atten: transpose then reshape)
// outMode 1: out[b, (n*31+c)*16384+d]   (nonlo: raw reshape, head-scrambled)
// ---------------------------------------------------------------------------
__global__ void k_attn_av(const float* __restrict__ att, const float* __restrict__ vb,
                          int vC, float* __restrict__ out, int outMode) {
  const int lane   = threadIdx.x & 31;
  const int laneHi = lane >> 4;
  const int nl     = lane & 15;
  const int wave   = blockIdx.x * (blockDim.x >> 5) + (threadIdx.x >> 5);
  const int mtile  = blockIdx.y;   // 0..1

  const int bn   = wave / 1024;    // (b*4 + head)
  const int dt   = wave % 1024;
  const int hd   = bn % 4;
  const int b    = bn / 4;
  const int dcol = dt * 16 + nl;
  const int c_a  = mtile * 16 + nl;
  const int rowA = (c_a < 31) ? c_a : 30;
  const float* bp = vb + (long)(b * vC + laneHi * 16) * HW + hd * 16384 + dcol;

  v8f acc = {0.f, 0.f, 0.f, 0.f, 0.f, 0.f, 0.f, 0.f};
  v16bf av, bv;
#pragma unroll
  for (int e = 0; e < 16; ++e) {
    const int kkA = e + laneHi * 16;
    const int kA  = (kkA < 31) ? kkA : 30;
    const float aval = att[((long)bn * 31 + rowA) * 31 + kA];
    av[e] = (__bf16)((kkA < 31 && c_a < 31) ? aval : 0.0f);
    const int kkB = e + laneHi * 16;
    const long boff = (kkB < 31) ? (long)e * HW : 0;   // clamp: reuse e=0 row
    const float bval = bp[boff];
    bv[e] = (__bf16)((kkB < 31) ? bval : 0.0f);
  }
  acc = wmma_bf16(av, bv, acc);
#pragma unroll
  for (int r = 0; r < 8; ++r) {
    const int c = mtile * 16 + laneHi * 8 + r;
    if (c < 31) {
      long off;
      if (outMode == 0) off = (long)(b * 31 + c) * HW + hd * 16384 + dcol;
      else              off = (long)b * 31 * HW + (long)(hd * 31 + c) * 16384 + dcol;
      out[off] = acc[r];
    }
  }
}

// --------------------------- VALU kernels ----------------------------------

template <int C>
__global__ void k_ln(const float* __restrict__ in, const float* __restrict__ w,
                     const float* __restrict__ bb, float* __restrict__ out) {
  const long p = (long)blockIdx.x * blockDim.x + threadIdx.x;
  if (p >= (long)NBATCH * HW) return;
  const int b = (int)(p / HW);
  const int hw = (int)(p % HW);
  const long base = (long)b * C * HW + hw;
  float vals[C];
  float mu = 0.f;
#pragma unroll
  for (int c = 0; c < C; ++c) { vals[c] = in[base + (long)c * HW]; mu += vals[c]; }
  mu /= (float)C;
  float var = 0.f;
#pragma unroll
  for (int c = 0; c < C; ++c) { const float d = vals[c] - mu; var += d * d; }
  var /= (float)C;
  const float rs = rsqrtf(var + 1e-5f);
#pragma unroll
  for (int c = 0; c < C; ++c)
    out[base + (long)c * HW] = (vals[c] - mu) * rs * w[c] + bb[c];
}

__global__ void k_dwconv3x3(const float* __restrict__ in, const float* __restrict__ Wt,
                            const float* __restrict__ bias, float* __restrict__ out,
                            int C, int act) {
  const long i = (long)blockIdx.x * blockDim.x + threadIdx.x;
  if (i >= (long)NBATCH * C * HW) return;
  const int b = (int)(i / ((long)C * HW));
  const int r = (int)(i % ((long)C * HW));
  const int c = r / HW;
  const int hw = r % HW;
  const int y = hw / IMG, x = hw % IMG;
  const float* ip = in + (long)(b * C + c) * HW;
  float s = bias ? bias[c] : 0.f;
#pragma unroll
  for (int ky = 0; ky < 3; ++ky) {
    const int yy = y + ky - 1;
#pragma unroll
    for (int kx = 0; kx < 3; ++kx) {
      const int xx = x + kx - 1;
      const bool ok = (yy >= 0) & (yy < IMG) & (xx >= 0) & (xx < IMG);
      const int yyc = (yy < 0) ? 0 : ((yy > IMG - 1) ? IMG - 1 : yy);
      const int xxc = (xx < 0) ? 0 : ((xx > IMG - 1) ? IMG - 1 : xx);
      const float v = ip[yyc * IMG + xxc];
      s += (ok ? v : 0.0f) * Wt[c * 9 + ky * 3 + kx];
    }
  }
  out[i] = act_apply(s, act);
}

__global__ void k_conv3x3_1to32(const float* __restrict__ in, const float* __restrict__ Wt,
                                const float* __restrict__ bias, float* __restrict__ out) {
  const long i = (long)blockIdx.x * blockDim.x + threadIdx.x;
  if (i >= (long)NBATCH * 32 * HW) return;
  const int b = (int)(i / (32l * HW));
  const int r = (int)(i % (32l * HW));
  const int oc = r / HW;
  const int hw = r % HW;
  const int y = hw / IMG, x = hw % IMG;
  const float* ip = in + (long)b * HW;
  float s = bias[oc];
#pragma unroll
  for (int ky = 0; ky < 3; ++ky) {
    const int yy = y + ky - 1;
#pragma unroll
    for (int kx = 0; kx < 3; ++kx) {
      const int xx = x + kx - 1;
      const bool ok = (yy >= 0) & (yy < IMG) & (xx >= 0) & (xx < IMG);
      const int yyc = (yy < 0) ? 0 : ((yy > IMG - 1) ? IMG - 1 : yy);
      const int xxc = (xx < 0) ? 0 : ((xx > IMG - 1) ? IMG - 1 : xx);
      const float v = ip[yyc * IMG + xxc];
      s += (ok ? v : 0.0f) * Wt[oc * 9 + ky * 3 + kx];
    }
  }
  out[i] = s;
}

__global__ void k_conv3x3_32to1(const float* __restrict__ in, const float* __restrict__ Wt,
                                const float* __restrict__ bias, float* __restrict__ out) {
  const long i = (long)blockIdx.x * blockDim.x + threadIdx.x;
  if (i >= (long)NBATCH * HW) return;
  const int b = (int)(i / HW);
  const int hw = (int)(i % HW);
  const int y = hw / IMG, x = hw % IMG;
  float s = bias[0];
  for (int ic = 0; ic < 32; ++ic) {
    const float* ip = in + (long)(b * 32 + ic) * HW;
#pragma unroll
    for (int ky = 0; ky < 3; ++ky) {
      const int yy = y + ky - 1;
#pragma unroll
      for (int kx = 0; kx < 3; ++kx) {
        const int xx = x + kx - 1;
        const bool ok = (yy >= 0) & (yy < IMG) & (xx >= 0) & (xx < IMG);
        const int yyc = (yy < 0) ? 0 : ((yy > IMG - 1) ? IMG - 1 : yy);
        const int xxc = (xx < 0) ? 0 : ((xx > IMG - 1) ? IMG - 1 : xx);
        const float v = ip[yyc * IMG + xxc];
        s += (ok ? v : 0.0f) * Wt[ic * 9 + ky * 3 + kx];
      }
    }
  }
  out[i] = s;
}

// per-(b,head,channel) L2 norm over d=16384
__global__ void k_rownorm(const float* __restrict__ base, int Ctot, int c0,
                          float* __restrict__ nrm) {
  const int id = blockIdx.x;            // ((b*4+n)*31 + c)
  const int c = id % 31;
  const int bn = id / 31;
  const int hd = bn % 4, b = bn / 4;
  const float* row = base + (long)(b * Ctot + c0 + c) * HW + hd * 16384;
  float s = 0.f;
  for (int d = threadIdx.x; d < 16384; d += 256) { const float v = row[d]; s += v * v; }
  __shared__ float sm[256];
  sm[threadIdx.x] = s; __syncthreads();
  for (int st = 128; st > 0; st >>= 1) {
    if (threadIdx.x < st) sm[threadIdx.x] += sm[threadIdx.x + st];
    __syncthreads();
  }
  if (threadIdx.x == 0) nrm[id] = sqrtf(sm[0]);
}

// logits[(bn,cq,ck)] = <q_row, k_row>  (d=16384 contraction)
__global__ void k_qk(const float* __restrict__ qb, int qC, int q0,
                     const float* __restrict__ kb, int kC, int k0,
                     float* __restrict__ logits) {
  const int id = blockIdx.x;            // ((bn*31+cq)*31+ck)
  const int ck = id % 31;
  const int t = id / 31;
  const int cq = t % 31;
  const int bn = t / 31;
  const int hd = bn % 4, b = bn / 4;
  const float* qr = qb + (long)(b * qC + q0 + cq) * HW + hd * 16384;
  const float* kr = kb + (long)(b * kC + k0 + ck) * HW + hd * 16384;
  float s = 0.f;
  for (int d = threadIdx.x; d < 16384; d += 256) s += qr[d] * kr[d];
  __shared__ float sm[256];
  sm[threadIdx.x] = s; __syncthreads();
  for (int st = 128; st > 0; st >>= 1) {
    if (threadIdx.x < st) sm[threadIdx.x] += sm[threadIdx.x + st];
    __syncthreads();
  }
  if (threadIdx.x == 0) logits[id] = sm[0];
}

__global__ void k_softmax31(const float* __restrict__ logits, const float* __restrict__ nq,
                            const float* __restrict__ nk, float* __restrict__ att) {
  const int id = blockIdx.x * blockDim.x + threadIdx.x;   // (bn*31 + cq)
  if (id >= NBATCH * 4 * 31) return;
  const int bn = id / 31;
  const float qn = fmaxf(nq[id], 1e-12f);
  float l[31], mx = -3.4e38f;
#pragma unroll
  for (int ck = 0; ck < 31; ++ck) {
    const float v = logits[(long)id * 31 + ck] / (qn * fmaxf(nk[bn * 31 + ck], 1e-12f));
    l[ck] = v; mx = fmaxf(mx, v);
  }
  float sum = 0.f;
#pragma unroll
  for (int ck = 0; ck < 31; ++ck) { l[ck] = __expf(l[ck] - mx); sum += l[ck]; }
  const float inv = 1.0f / sum;
#pragma unroll
  for (int ck = 0; ck < 31; ++ck) att[(long)id * 31 + ck] = l[ck] * inv;
}

// out31 = w1*A[:,a0:a0+31] + w2*B[:,b0:b0+31]
__global__ void k_vmix(const float* __restrict__ A, int AC, int a0,
                       const float* __restrict__ Bs, int BC, int b0,
                       const float* __restrict__ w1p, const float* __restrict__ w2p,
                       float* __restrict__ out) {
  const long i = (long)blockIdx.x * blockDim.x + threadIdx.x;
  if (i >= (long)NBATCH * 31 * HW) return;
  const int b = (int)(i / (31l * HW));
  const int r = (int)(i % (31l * HW));
  const int c = r / HW;
  const int hw = r % HW;
  out[i] = w1p[0] * A[(long)(b * AC + a0 + c) * HW + hw] +
           w2p[0] * Bs[(long)(b * BC + b0 + c) * HW + hw];
}

__global__ void k_lerp(const float* __restrict__ a, const float* __restrict__ bb,
                       const float* __restrict__ lam, float* __restrict__ out, long n) {
  const long i = (long)blockIdx.x * blockDim.x + threadIdx.x;
  if (i < n) out[i] = a[i] + lam[0] * (bb[i] - a[i]);
}

__global__ void k_add2(const float* __restrict__ a, const float* __restrict__ bb,
                       float* __restrict__ out, long n) {
  const long i = (long)blockIdx.x * blockDim.x + threadIdx.x;
  if (i < n) out[i] = a[i] + bb[i];
}

__global__ void k_add3(const float* __restrict__ a, const float* __restrict__ bb,
                       const float* __restrict__ cc, float* __restrict__ out, long n) {
  const long i = (long)blockIdx.x * blockDim.x + threadIdx.x;
  if (i < n) out[i] = a[i] + bb[i] + cc[i];
}

__global__ void k_copychan(const float* __restrict__ src, int sC, int sc0,
                           float* __restrict__ dst, int dC, int dc0, int nch) {
  const long i = (long)blockIdx.x * blockDim.x + threadIdx.x;
  if (i >= (long)NBATCH * nch * HW) return;
  const int b = (int)(i / ((long)nch * HW));
  const int r = (int)(i % ((long)nch * HW));
  const int c = r / HW;
  const int hw = r % HW;
  dst[(long)(b * dC + dc0 + c) * HW + hw] = src[(long)(b * sC + sc0 + c) * HW + hw];
}

// ----------------------------- host side -----------------------------------

// input indices (recursive dict order of setup_inputs())
enum {
  IN_X = 0, IN_ZPRE, IN_ZCUR, IN_PHITB,
  A_N1W, A_N1B, A_N2W, A_N2B, A_QV1W1, A_QV1B1, A_QV1W2, A_QV1B2,
  A_KVW1, A_KVB1, A_KVW2, A_KVB2, A_OUTW, A_OUTB, A_W1, A_W2,
  N_NXW, N_NXB, N_NZW, N_NZB, N_TW1, N_TB1, N_TW2, N_TB2,
  N_PW1, N_PB1, N_PW2, N_PB2, N_G1W1, N_G1B1, N_G1W2, N_G1B2,
  N_G2W1, N_G2B1, N_G2W2, N_G2B2, N_WW, N_WB, N_VW, N_VB,
  N_PEW1, N_PEW2, N_W3, N_W4,
  P_LAMBDA, P_CEW, P_CEB, P_GMW1, P_GMB1, P_GMW2, P_GMB2,
  P_N1W, P_N1B, P_N2W, P_N2B, P_CFW1, P_CFW2, P_CFW3,
  P_CBW1, P_CBW2, P_CBW3, P_CCW, P_CCB
};

static inline long cdivl(long a, long b) { return (a + b - 1) / b; }

extern "C" void kernel_launch(void* const* d_in, const int* in_sizes, int n_in,
                              void* d_out, int out_size, void* d_ws, size_t ws_size,
                              hipStream_t stream) {
  (void)in_sizes; (void)n_in; (void)out_size; (void)ws_size;
  auto IN = [&](int i) { return (const float*)d_in[i]; };
  float* P = (float*)d_ws;
  auto S  = [&](int i) { return P + (size_t)i * SLOT; };

  float* nq     = P + 11ul * SLOT;
  float* nk     = nq + 992;
  float* logits = nk + 992;
  float* att    = logits + 30752;

  const dim3 blk(256);
  const unsigned gx = NBATCH * HW / 16 / 8;   // waves = pixel tiles
  const long n32 = (long)NBATCH * 32 * HW;
  const long n31 = (long)NBATCH * 31 * HW;
  const dim3 gPix((unsigned)cdivl((long)NBATCH * HW, 256));

  auto conv1x1 = [&](const float* in, const float* Wt, const float* bias,
                     const float* res, float* out, int Cin, int Cout, int act) {
    dim3 g(gx, (unsigned)((Cout + 15) / 16));
    k_conv1x1<<<g, blk, 0, stream>>>(in, Wt, bias, res, out, Cin, Cout, act);
  };
  auto dwconv = [&](const float* in, const float* Wt, const float* bias,
                    float* out, int C, int act) {
    dim3 g((unsigned)cdivl((long)NBATCH * C * HW, 256));
    k_dwconv3x3<<<g, blk, 0, stream>>>(in, Wt, bias, out, C, act);
  };

  // ===================== Phase 1: z = atten(z_pre, z_cur) ====================
  k_ln<31><<<gPix, blk, 0, stream>>>(IN(IN_ZPRE), IN(A_N1W), IN(A_N1B), S(0)); // pre_ln
  k_ln<31><<<gPix, blk, 0, stream>>>(IN(IN_ZCUR), IN(A_N2W), IN(A_N2B), S(1)); // cur_ln
  conv1x1(S(1), IN(A_QV1W1), IN(A_QV1B1), nullptr, S(2), 31, 62, 0);
  dwconv(S(2), IN(A_QV1W2), IN(A_QV1B2), S(4), 62, 0);             // qv1 -> s4..5
  conv1x1(S(0), IN(A_KVW1), IN(A_KVB1), nullptr, S(2), 31, 62, 0);
  dwconv(S(2), IN(A_KVW2), IN(A_KVB2), S(6), 62, 0);               // kv -> s6..7
  {
    dim3 g((unsigned)cdivl(n31, 256));
    k_vmix<<<g, blk, 0, stream>>>(S(4), 62, 31, S(6), 62, 31, IN(A_W1), IN(A_W2), S(8));
  }
  k_rownorm<<<dim3(992), blk, 0, stream>>>(S(4), 62, 0, nq);
  k_rownorm<<<dim3(992), blk, 0, stream>>>(S(6), 62, 0, nk);
  k_qk<<<dim3(30752), blk, 0, stream>>>(S(4), 62, 0, S(6), 62, 0, logits);
  k_softmax31<<<dim3(4), blk, 0, stream>>>(logits, nq, nk, att);
  k_attn_av<<<dim3(gx, 2), blk, 0, stream>>>(att, S(8), 31, S(0), 0);
  conv1x1(S(0), IN(A_OUTW), IN(A_OUTB), IN(IN_ZCUR), S(9), 31, 31, 0);   // z -> s9

  // ===================== Phase 2: main pipeline =============================
  {
    dim3 g((unsigned)cdivl(n32, 256));
    k_conv3x3_1to32<<<g, blk, 0, stream>>>(IN(IN_X), IN(P_CEW), IN(P_CEB), S(0));
    k_conv3x3_1to32<<<g, blk, 0, stream>>>(IN(IN_PHITB), IN(P_CEW), IN(P_CEB), S(1));
    k_lerp<<<g, blk, 0, stream>>>(S(0), S(1), IN(P_LAMBDA), S(2), n32);  // x_grad
  }
  k_conv3x3_32<<<dim3(gx, 2), blk, 0, stream>>>(S(2), IN(P_GMW1), IN(P_GMB1), S(0), 1);
  k_conv3x3_32<<<dim3(gx, 2), blk, 0, stream>>>(S(0), IN(P_GMW2), IN(P_GMB2), S(1), 0);
  k_add2<<<dim3((unsigned)cdivl(n32, 256)), blk, 0, stream>>>(S(2), S(1), S(3), n32);

  // ---- nonlo(x=s3, z=s9) ----
  k_ln<32><<<gPix, blk, 0, stream>>>(S(3), IN(N_NXW), IN(N_NXB), S(0));  // x0
  k_ln<31><<<gPix, blk, 0, stream>>>(S(9), IN(N_NZW), IN(N_NZB), S(1));  // z0
  conv1x1(S(1), IN(N_TW1), IN(N_TB1), nullptr, S(2), 31, 31, 0);
  dwconv(S(2), IN(N_TW2), IN(N_TB2), S(4), 31, 0);                  // z1 -> s4
  conv1x1(S(0), IN(N_PW1), IN(N_PB1), nullptr, S(2), 32, 31, 0);
  dwconv(S(2), IN(N_PW2), IN(N_PB2), S(5), 31, 0);                  // x1 -> s5
  conv1x1(S(0), IN(N_G1W1), IN(N_G1B1), nullptr, S(2), 32, 31, 0);
  dwconv(S(2), IN(N_G1W2), IN(N_G1B2), S(6), 31, 0);                // x2 -> s6
  conv1x1(S(1), IN(N_G2W1), IN(N_G2B1), nullptr, S(2), 31, 31, 0);
  dwconv(S(2), IN(N_G2W2), IN(N_G2B2), S(7), 31, 0);                // z2 -> s7
  {
    dim3 g((unsigned)cdivl(n31, 256));
    k_vmix<<<g, blk, 0, stream>>>(S(7), 31, 0, S(6), 31, 0, IN(N_W3), IN(N_W4), S(8));
  }
  k_rownorm<<<dim3(992), blk, 0, stream>>>(S(4), 31, 0, nq);   // |z1|
  k_rownorm<<<dim3(992), blk, 0, stream>>>(S(5), 31, 0, nk);   // |x1|
  k_qk<<<dim3(30752), blk, 0, stream>>>(S(4), 31, 0, S(5), 31, 0, logits);
  k_softmax31<<<dim3(4), blk, 0, stream>>>(logits, nq, nk, att);
  k_attn_av<<<dim3(gx, 2), blk, 0, stream>>>(att, S(8), 31, S(5), 1);
  dwconv(S(7), IN(N_PEW1), nullptr, S(4), 31, 2);                   // gelu(dw(z2))
  dwconv(S(4), IN(N_PEW2), nullptr, S(2), 31, 0);                   // pos -> s2
  conv1x1(S(5), IN(N_WW), IN(N_WB), nullptr, S(7), 31, 31, 0);      // wout -> s7
  k_add3<<<dim3((unsigned)cdivl(n31, 256)), blk, 0, stream>>>(S(7), S(2), S(9), S(4), n31);
  {                                                                 // concat -> X63 @ s6..7
    dim3 g32((unsigned)cdivl(n32, 256)), g31((unsigned)cdivl(n31, 256));
    k_copychan<<<g32, blk, 0, stream>>>(S(3), 32, 0, S(6), 63, 0, 32);
    k_copychan<<<g31, blk, 0, stream>>>(S(4), 31, 0, S(6), 63, 32, 31);
  }
  conv1x1(S(6), IN(N_VW), IN(N_VB), nullptr, S(0), 63, 32, 0);      // x_input -> s0

  // ---- FFN 1 ----
  k_ln<32><<<gPix, blk, 0, stream>>>(S(0), IN(P_N1W), IN(P_N1B), S(1));   // xn
  conv1x1(S(1), IN(P_CFW1), nullptr, nullptr, S(2), 32, 128, 2);    // gelu, F1 -> s2..5
  dwconv(S(2), IN(P_CFW2), nullptr, S(6), 128, 2);                  // gelu, F2 -> s6..9
  conv1x1(S(6), IN(P_CFW3), nullptr, S(0), S(1), 128, 32, 0);       // x_fwd -> s1

  // ---- FFN 2 ----
  k_ln<32><<<gPix, blk, 0, stream>>>(S(1), IN(P_N2W), IN(P_N2B), S(2));   // xn2
  conv1x1(S(2), IN(P_CBW1), nullptr, nullptr, S(3), 32, 128, 2);    // F1b -> s3..6
  dwconv(S(3), IN(P_CBW2), nullptr, S(7), 128, 2);                  // F2b -> s7..10
  conv1x1(S(7), IN(P_CBW3), nullptr, S(1), S(2), 128, 32, 0);       // x_bwd -> s2

  k_add2<<<dim3((unsigned)cdivl(n32, 256)), blk, 0, stream>>>(S(0), S(2), S(3), n32);

  float* outF = (float*)d_out;
  k_conv3x3_32to1<<<dim3((unsigned)cdivl((long)NBATCH * HW, 256)), blk, 0, stream>>>(
      S(3), IN(P_CCW), IN(P_CCB), outF);
  k_copychan<<<dim3((unsigned)cdivl(n31, 256)), blk, 0, stream>>>(
      S(3), 32, 0, outF + (long)NBATCH * HW, 31, 0, 31);
}
